// CountingModel_81441169866772
// MI455X (gfx1250) — matrix operand order repaired and checked by reading, not compiled, not run on hardware
//
#include <hip/hip_runtime.h>
#include <hip/hip_bf16.h>

// ---------------------------------------------------------------------------
// SRU stack on MI455X (gfx1250): bf16 WMMA GEMMs + chunked parallel scan.
// ---------------------------------------------------------------------------

#define USE_ASYNC_LDS 1

typedef unsigned short u16;
typedef __attribute__((ext_vector_type(16))) __bf16 v16bf;
typedef __attribute__((ext_vector_type(8)))  float  v8f;

constexpr int L_SEQ   = 2048;
constexpr int BATCH   = 16;
constexpr int HID     = 512;
constexpr int OUTN    = 10;
constexpr int NLAYERS = 5;
constexpr int M_TOT   = L_SEQ * BATCH;   // 32768
constexpr int N3H     = 3 * HID;         // 1536
constexpr int NCH     = BATCH * HID;     // 8192 channels for the scan
constexpr int CHUNKS  = 16;
constexpr int CLEN    = L_SEQ / CHUNKS;  // 128

// LDS tile row stride (in u16) for a 128x64 bf16 tile: 16B-aligned rows,
// stride 36 dwords -> conflict-friendly on 64 banks.
#define KSTR 72
constexpr unsigned TILE_BYTES = 128 * KSTR * 2;  // 18432 per buffer

#if USE_ASYNC_LDS
#define WAIT_ASYNC() asm volatile("s_wait_asynccnt 0" ::: "memory")
#else
#define WAIT_ASYNC()
#endif

__device__ __forceinline__ u16 f2bf(float f) {
    unsigned u = __float_as_uint(f);
    unsigned r = u + 0x7FFFu + ((u >> 16) & 1u);   // round-to-nearest-even
    return (u16)(r >> 16);
}

__device__ __forceinline__ float sigmoidf(float z) {
    return 1.0f / (1.0f + __expf(-z));
}

// ---------------------------------------------------------------------------
// f32 -> bf16 conversion (grid-stride)
// ---------------------------------------------------------------------------
__global__ void f32_to_bf16_kernel(const float* __restrict__ in,
                                   u16* __restrict__ out, int n) {
    int i = blockIdx.x * blockDim.x + threadIdx.x;
    int stride = gridDim.x * blockDim.x;
    for (; i < n; i += stride) out[i] = f2bf(in[i]);
}

// ---------------------------------------------------------------------------
// GEMM: C[M x 1536] = A[M x 512](bf16) * B[1536 x 512](bf16, row-major)^T
// Block tile 128x128, BK=64, 256 threads = 8 wave32 waves (4 M-groups x 2 N).
// Each wave: 2x4 grid of 16x16 f32 accumulators via v_wmma_f32_16x16x32_bf16.
// Tiles staged through LDS with async global->LDS copies (GVS saddr form:
// per-lane 32-bit voffset is K-invariant; K advance is a scalar base).
// ---------------------------------------------------------------------------
union Frag { v16bf v; uint4 q[2]; };
union Acc  { v8f v; float f[8]; };

__global__ __launch_bounds__(256) void gemm_bf16_kernel(
        const u16* __restrict__ A, const u16* __restrict__ Bw,
        float* __restrict__ C) {
    __shared__ u16 sA[2][128 * KSTR];
    __shared__ u16 sB[2][128 * KSTR];

    const int tid  = threadIdx.x;
    const int lane = tid & 31;
    const int wave = tid >> 5;
    const int wm   = wave >> 1;            // 0..3 -> M offset wm*32
    const int wn   = wave & 1;             // 0..1 -> N offset wn*64
    const size_t m0 = (size_t)blockIdx.x * 128;
    const int    n0 = blockIdx.y * 128;

    const int lrow   = lane & 15;
    const int lhi    = lane >> 4;          // 0/1
    const int kbaseA = lhi << 3;           // A: K block 0 / 8  (then +16)
    const int kbaseB = lhi << 3;           // B: K block 0 / 8  (then +16)

    // Per-lane copy offsets (K-invariant, bytes).
    unsigned voff[4], relo[4];
#pragma unroll
    for (int i = 0; i < 4; ++i) {
        int idx = tid + i * 256;           // uint4 index 0..1023
        int row = idx >> 3;                // 8 uint4 per row of 64 bf16
        int kc  = (idx & 7) << 3;
        voff[i] = (unsigned)((row * 512 + kc) * 2);   // global byte offset
        relo[i] = (unsigned)((row * KSTR + kc) * 2);  // LDS byte offset
    }
    const unsigned ldsA0 = (unsigned)(unsigned long long)&sA[0][0];
    const unsigned ldsB0 = (unsigned)(unsigned long long)&sB[0][0];
    const unsigned long long gA = (unsigned long long)(A  + m0 * 512);
    const unsigned long long gB = (unsigned long long)(Bw + (size_t)n0 * 512);

    auto issue_tiles = [&](int buf, int kbytes) {
        const unsigned long long ga = gA + (unsigned)kbytes;
        const unsigned long long gb = gB + (unsigned)kbytes;
        const unsigned la = ldsA0 + (unsigned)buf * TILE_BYTES;
        const unsigned lb = ldsB0 + (unsigned)buf * TILE_BYTES;
#pragma unroll
        for (int i = 0; i < 4; ++i) {
#if USE_ASYNC_LDS
            asm volatile("global_load_async_to_lds_b128 %0, %1, %2"
                         :: "v"(la + relo[i]), "v"(voff[i]), "s"(ga) : "memory");
            asm volatile("global_load_async_to_lds_b128 %0, %1, %2"
                         :: "v"(lb + relo[i]), "v"(voff[i]), "s"(gb) : "memory");
#else
            *(uint4*)((char*)&sA[buf][0] + relo[i]) =
                *(const uint4*)((const char*)gA + kbytes + voff[i]);
            *(uint4*)((char*)&sB[buf][0] + relo[i]) =
                *(const uint4*)((const char*)gB + kbytes + voff[i]);
#endif
        }
    };

    Acc acc[2][4];
#pragma unroll
    for (int mi = 0; mi < 2; ++mi)
#pragma unroll
        for (int ni = 0; ni < 4; ++ni)
#pragma unroll
            for (int i = 0; i < 8; ++i) acc[mi][ni].f[i] = 0.0f;

    issue_tiles(0, 0);
    WAIT_ASYNC();
    __syncthreads();

#pragma unroll
    for (int kt = 0; kt < 8; ++kt) {          // K = 512 in steps of 64
        const int cur = kt & 1;
        if (kt < 7) issue_tiles(cur ^ 1, (kt + 1) * 128);  // bytes: 64 cols * 2

        const u16* As = &sA[cur][0];
        const u16* Bs = &sB[cur][0];
#pragma unroll
        for (int ks = 0; ks < 64; ks += 32) {
            Frag af[2], bfr[4];
#pragma unroll
            for (int mi = 0; mi < 2; ++mi) {
                const u16* p = As + (wm * 32 + mi * 16 + lrow) * KSTR + ks + kbaseA;
                af[mi].q[0] = *(const uint4*)p;         // K = kbase + 0..7
                af[mi].q[1] = *(const uint4*)(p + 16);  // K = kbase + 16..23
            }
#pragma unroll
            for (int ni = 0; ni < 4; ++ni) {
                const u16* p = Bs + (wn * 64 + ni * 16 + lrow) * KSTR + ks + kbaseB;
                bfr[ni].q[0] = *(const uint4*)p;         // K = kbase + 0..7
                bfr[ni].q[1] = *(const uint4*)(p + 16);  // K = kbase + 16..23
            }
#pragma unroll
            for (int mi = 0; mi < 2; ++mi)
#pragma unroll
                for (int ni = 0; ni < 4; ++ni)
                    acc[mi][ni].v = __builtin_amdgcn_wmma_f32_16x16x32_bf16(
                        false, af[mi].v, false, bfr[ni].v,
                        (short)0, acc[mi][ni].v, false, false);
        }
        if (kt < 7) { WAIT_ASYNC(); __syncthreads(); }
    }

    // D layout: VGPR i -> M = i (+8 for lanes 16..31), N = lane & 15.
#pragma unroll
    for (int mi = 0; mi < 2; ++mi) {
        const size_t rbase = m0 + wm * 32 + mi * 16 + (lhi ? 8 : 0);
#pragma unroll
        for (int ni = 0; ni < 4; ++ni) {
            const int col = n0 + wn * 64 + ni * 16 + lrow;
            float* p = C + rbase * (size_t)N3H + col;
#pragma unroll
            for (int i = 0; i < 8; ++i)
                p[i * N3H] = acc[mi][ni].f[i];   // imm offsets i*6144B
        }
    }
}

// ---------------------------------------------------------------------------
// SRU scan, chunked linear-recurrence (c_t = f*c + (1-f)*xt is linear in c).
// ---------------------------------------------------------------------------
// Pass 1: per (channel, chunk): f = sigmoid(zf+bf) (stored back over zf slot),
//         a = prod f, cend = local scan with c_in = 0.
__global__ __launch_bounds__(256) void sru_pass1(
        float* __restrict__ U, const float* __restrict__ bsl,
        float* __restrict__ Aar, float* __restrict__ Cend) {
    int gid   = blockIdx.x * 256 + threadIdx.x;  // chunk*8192 + ch
    int ch    = gid & (NCH - 1);
    int chunk = gid >> 13;
    int b = ch >> 9, hd = ch & (HID - 1);
    float bfv = bsl[hd];
    float a = 1.0f, c = 0.0f;
    size_t base = ((size_t)(chunk * CLEN) * BATCH + b) * N3H + hd;
    const size_t stride = (size_t)BATCH * N3H;
    for (int i = 0; i < CLEN; ++i) {
        float xt = U[base];
        float zf = U[base + HID];
        float f  = sigmoidf(zf + bfv);
        U[base + HID] = f;                // cache gate for pass 3
        a *= f;
        c = f * c + (1.0f - f) * xt;
        base += stride;
    }
    Aar[gid] = a;
    Cend[gid] = c;
}

// Pass 2: stitch carries across the 16 chunks of each channel.
__global__ __launch_bounds__(256) void sru_pass2(
        const float* __restrict__ Aar, const float* __restrict__ Cend,
        float* __restrict__ Cin) {
    int ch = blockIdx.x * 256 + threadIdx.x;     // 0..8191
    float c = 0.0f;
    for (int k = 0; k < CHUNKS; ++k) {
        Cin[k * NCH + ch] = c;
        c = Aar[k * NCH + ch] * c + Cend[k * NCH + ch];
    }
}

// Pass 3: replay with correct carry-in; highway merge; emit f32 + bf16 h.
__global__ __launch_bounds__(256) void sru_pass3(
        const float* __restrict__ U, const float* __restrict__ bsl,
        const float* __restrict__ Cin, const float* __restrict__ Xin,
        float* __restrict__ Hf, u16* __restrict__ Hb) {
    int gid   = blockIdx.x * 256 + threadIdx.x;
    int ch    = gid & (NCH - 1);
    int chunk = gid >> 13;
    int b = ch >> 9, hd = ch & (HID - 1);
    float brv = bsl[HID + hd];
    float c = Cin[gid];
    size_t base  = ((size_t)(chunk * CLEN) * BATCH + b) * N3H + hd;
    size_t xbase = ((size_t)(chunk * CLEN) * BATCH + b) * HID + hd;
    const size_t stride  = (size_t)BATCH * N3H;
    const size_t xstride = (size_t)BATCH * HID;
    for (int i = 0; i < CLEN; ++i) {
        float xt = U[base];
        float f  = U[base + HID];         // precomputed gate
        float zr = U[base + 2 * HID];
        c = f * c + (1.0f - f) * xt;
        float r  = sigmoidf(zr + brv);
        float xi = Xin[xbase];
        float h  = r * c + (1.0f - r) * xi;
        Hf[xbase] = h;                    // in-place safe: same thread/index
        Hb[xbase] = f2bf(h);
        base += stride;
        xbase += xstride;
    }
}

// ---------------------------------------------------------------------------
// Final FC: out[m, o] = sum_k H[m,k] * W[o,k] + b[o]  (N=10, tiny)
// ---------------------------------------------------------------------------
__global__ __launch_bounds__(256) void fc_kernel(
        const float* __restrict__ H, const float* __restrict__ W,
        const float* __restrict__ bias, float* __restrict__ out) {
    __shared__ float sh[64][65];
    __shared__ float sw[OUTN][65];
    const int tid = threadIdx.x;
    const int m0  = blockIdx.x * 64;
    float acc[3] = {0.f, 0.f, 0.f};
    for (int kc = 0; kc < HID; kc += 64) {
        __syncthreads();
#pragma unroll
        for (int i = 0; i < 16; ++i) {
            int fl = tid + i * 256;
            int r = fl >> 6, c = fl & 63;
            sh[r][c] = H[(size_t)(m0 + r) * HID + kc + c];
        }
        for (int fl = tid; fl < OUTN * 64; fl += 256) {
            int o = fl >> 6, c = fl & 63;
            sw[o][c] = W[o * HID + kc + c];
        }
        __syncthreads();
#pragma unroll
        for (int p = 0; p < 3; ++p) {
            int fl = tid + p * 256;
            if (fl < 64 * OUTN) {
                int r = fl / OUTN, o = fl % OUTN;
                float s = 0.f;
#pragma unroll
                for (int c = 0; c < 64; ++c) s += sh[r][c] * sw[o][c];
                acc[p] += s;
            }
        }
    }
#pragma unroll
    for (int p = 0; p < 3; ++p) {
        int fl = tid + p * 256;
        if (fl < 64 * OUTN) {
            int r = fl / OUTN, o = fl % OUTN;
            out[(size_t)(m0 + r) * OUTN + o] = acc[p] + bias[o];
        }
    }
}

// ---------------------------------------------------------------------------
extern "C" void kernel_launch(void* const* d_in, const int* in_sizes, int n_in,
                              void* d_out, int out_size, void* d_ws, size_t ws_size,
                              hipStream_t stream) {
    const float* x   = (const float*)d_in[0];   // [2048,16,512]
    const float* Ws  = (const float*)d_in[1];   // [5,1536,512]
    const float* bs  = (const float*)d_in[2];   // [5,1024]
    const float* fcW = (const float*)d_in[3];   // [10,512]
    const float* fcb = (const float*)d_in[4];   // [10]
    float* out = (float*)d_out;                 // [2048,16,10]

    char* ws = (char*)d_ws;
    size_t off = 0;
    auto carve = [&](size_t bytes) -> char* {
        char* p = ws + off;
        off += (bytes + 255) & ~(size_t)255;
        return p;
    };
    float* U   = (float*)carve((size_t)M_TOT * N3H * sizeof(float));       // 192 MB
    float* Hf  = (float*)carve((size_t)M_TOT * HID * sizeof(float));       //  64 MB
    u16*   Hb  = (u16*)  carve((size_t)M_TOT * HID * sizeof(u16));         //  32 MB
    u16*   Wb  = (u16*)  carve((size_t)NLAYERS * N3H * HID * sizeof(u16)); //   8 MB
    float* Aar = (float*)carve((size_t)CHUNKS * NCH * sizeof(float));
    float* Ce  = (float*)carve((size_t)CHUNKS * NCH * sizeof(float));
    float* Ci  = (float*)carve((size_t)CHUNKS * NCH * sizeof(float));
    (void)ws_size; (void)in_sizes; (void)n_in; (void)out_size;

    // bf16 copies of activations (layer-0 input) and all layer weights.
    f32_to_bf16_kernel<<<8192, 256, 0, stream>>>(x, Hb, M_TOT * HID);
    f32_to_bf16_kernel<<<8192, 256, 0, stream>>>(Ws, Wb, NLAYERS * N3H * HID);

    const dim3 ggrid(M_TOT / 128, N3H / 128);   // 256 x 12
    for (int l = 0; l < NLAYERS; ++l) {
        gemm_bf16_kernel<<<ggrid, 256, 0, stream>>>(
            Hb, Wb + (size_t)l * N3H * HID, U);
        sru_pass1<<<(NCH * CHUNKS) / 256, 256, 0, stream>>>(
            U, bs + l * 2 * HID, Aar, Ce);
        sru_pass2<<<NCH / 256, 256, 0, stream>>>(Aar, Ce, Ci);
        sru_pass3<<<(NCH * CHUNKS) / 256, 256, 0, stream>>>(
            U, bs + l * 2 * HID, Ci, (l == 0 ? x : Hf), Hf, Hb);
    }
    fc_kernel<<<M_TOT / 64, 256, 0, stream>>>(Hf, fcW, fcb, out);
}
